// CPMAttention_38216619000109
// MI455X (gfx1250) — matrix operand order repaired and checked by reading, not compiled, not run on hardware
//
#include <hip/hip_runtime.h>
#include <hip/hip_bf16.h>

// ---------------------------------------------------------------------------
// CDNA5 / gfx1250: all matmuls on v_wmma_f32_16x16x32_f16 (wave32 WMMA),
// double-buffered GLOBAL_LOAD_ASYNC_TO_LDS_B128 staging (ASYNCcnt pipelining),
// operand-reuse hints on shared-A WMMA chains, flash attention w/ online softmax.
// ---------------------------------------------------------------------------

typedef _Float16 v16h __attribute__((ext_vector_type(16)));
typedef _Float16 v8h  __attribute__((ext_vector_type(8)));
typedef float    v8f  __attribute__((ext_vector_type(8)));

union HU { v16h v; v8h h[2]; };

#define DEVINL __device__ __forceinline__

// Problem constants
static constexpr int Bc   = 2;
static constexpr int Sc   = 1024;
static constexpr int Hc   = 4096;
static constexpr int NHc  = 32;
static constexpr int NKVc = 8;
static constexpr int HDc  = 128;
static constexpr int QKVN = 6144;           // 4096 + 1024 + 1024
static constexpr float SCALEc = 0.08838834764831845f; // 128^-0.5

// WMMA wrapper: reuse-A hint must be an integer constant -> template param.
template <bool RA>
DEVINL v8f wmma_f16(v16h a, v16h b, v8f c) {
  return __builtin_amdgcn_wmma_f32_16x16x32_f16(false, a, false, b,
                                                (short)0, c, RA, false);
}

// ---------------------------------------------------------------------------
// CDNA5 async copy: global -> LDS, ASYNCcnt-tracked (no VGPR round-trip).
// LDS address VGPR = low 32 bits of the generic pointer (ISA aperture rule:
// address truncated to its low 32 bits).
// ---------------------------------------------------------------------------
DEVINL void async_ld32B(unsigned lds_off, const void* gaddr) {
  unsigned long long g = (unsigned long long)(uintptr_t)gaddr;
  asm volatile(
      "global_load_async_to_lds_b128 %0, %1, off\n\t"
      "global_load_async_to_lds_b128 %0, %1, off offset:16"
      :: "v"(lds_off), "v"(g) : "memory");
}
DEVINL void async_ld64B(unsigned lds_off, const void* gaddr) {
  unsigned long long g = (unsigned long long)(uintptr_t)gaddr;
  asm volatile(
      "global_load_async_to_lds_b128 %0, %1, off\n\t"
      "global_load_async_to_lds_b128 %0, %1, off offset:16\n\t"
      "global_load_async_to_lds_b128 %0, %1, off offset:32\n\t"
      "global_load_async_to_lds_b128 %0, %1, off offset:48"
      :: "v"(lds_off), "v"(g) : "memory");
}
DEVINL void wait_async0() {
  asm volatile("s_wait_asynccnt 0x0" ::: "memory");
}
DEVINL unsigned lds_addr(const void* p) {
  return (unsigned)(uintptr_t)p;  // low 32 bits = LDS byte offset
}

// ---------------------------------------------------------------------------
// Fragment loaders following the ISA wave32 VGPR layouts.
// A 16x32 f16: lane m = lane&15; lanes<16 hold K {0..7,16..23},
//              lanes>=16 hold K {8..15,24..31}  -> two contiguous 16B runs.
// B 32x16 f16 stored as Bt[n][k]: lane n = lane&15; lanes<16 K {0..7,16..23},
//              lanes>=16 K {8..15,24..31} -> two contiguous 16B runs.
// C/D 16x16 f32: lane n = lane&15, VGPR v -> m = v + (lane>=16 ? 8 : 0).
// ---------------------------------------------------------------------------
DEVINL v16h load_afrag(const _Float16* base, int rs, int lane) {
  const _Float16* r = base + (size_t)(lane & 15) * rs + (((lane >> 4) & 1) * 8);
  HU u;
  u.h[0] = *(const v8h*)(r);
  u.h[1] = *(const v8h*)(r + 16);
  return u.v;
}
DEVINL v16h load_bfrag(const _Float16* base, int rs, int lane) {
  const _Float16* r = base + (size_t)(lane & 15) * rs + (((lane >> 4) & 1) * 8);
  HU u;
  u.h[0] = *(const v8h*)(r);
  u.h[1] = *(const v8h*)(r + 16);
  return u.v;
}

// ---------------------------------------------------------------------------
// fp32 -> fp16 conversion (4 elements / thread)
// ---------------------------------------------------------------------------
__global__ __launch_bounds__(256) void cvt_f32_f16(const float* __restrict__ in,
                                                   _Float16* __restrict__ out,
                                                   int n) {
  int i = (blockIdx.x * 256 + threadIdx.x) * 4;
  if (i >= n) return;
  float4 v = *(const float4*)(in + i);
  union { _Float16 h[4]; uint2 u; } t;
  t.h[0] = (_Float16)v.x; t.h[1] = (_Float16)v.y;
  t.h[2] = (_Float16)v.z; t.h[3] = (_Float16)v.w;
  *(uint2*)(out + i) = t.u;
}

// ---------------------------------------------------------------------------
// fp32 [K][N] -> fp16 [N][K] tiled transpose-convert (for weight matrices):
// makes GEMM B-tile staging a pure 128-bit async copy (no LDS scatter).
// ---------------------------------------------------------------------------
__global__ __launch_bounds__(256)
void transpose_cvt_f32_f16(const float* __restrict__ in,
                           _Float16* __restrict__ out, int K, int N) {
  __shared__ float tile[32][33];
  const int n0 = blockIdx.x * 32, k0 = blockIdx.y * 32;
  const int tx = threadIdx.x & 31, ty = threadIdx.x >> 5;  // 32 x 8
#pragma unroll
  for (int r = 0; r < 32; r += 8)
    tile[ty + r][tx] = in[(size_t)(k0 + ty + r) * N + n0 + tx];
  __syncthreads();
#pragma unroll
  for (int r = 0; r < 32; r += 8)
    out[(size_t)(n0 + ty + r) * K + k0 + tx] = (_Float16)tile[tx][ty + r];
}

// ---------------------------------------------------------------------------
// WMMA GEMM: C[M,N] (f32) = A[M,K] (f16, row-major) * Bt[N,K] (f16, row-major)
// Block tile 128x256, K-step 32, 8 waves (2x4), wave tile 64x64 = 4x4 WMMA
// tiles -> 16 WMMAs per 8 fragment loads per chunk (2x LDS-read efficiency).
// Double-buffered async-to-LDS staging pipelined on ASYNCcnt.
// ---------------------------------------------------------------------------
__global__ __launch_bounds__(256)
void gemm_f16_wmma(const _Float16* __restrict__ A,
                   const _Float16* __restrict__ Bt,
                   float* __restrict__ C,
                   int M, int N, int K) {
  __shared__ alignas(16) _Float16 lA[2][128][32];   // 16 KB
  __shared__ alignas(16) _Float16 lB[2][256][32];   // 32 KB

  const int tid  = threadIdx.x;
  const int lane = tid & 31;
  const int wave = tid >> 5;
  const int tile_m = blockIdx.y * 128;
  const int tile_n = blockIdx.x * 256;
  const int wm = (wave >> 2) * 64;  // 0 or 64
  const int wn = (wave & 3) * 64;   // 0,64,128,192

  // staging: A 32B/thread (2 threads/row), B 64B/thread (1 thread/row)
  const int ar = tid >> 1, ac = (tid & 1) * 16;
  const int br = tid;
  const _Float16* ga = A  + (size_t)(tile_m + ar) * K + ac;
  const _Float16* gb = Bt + (size_t)(tile_n + br) * K;
  const unsigned la[2] = { lds_addr(&lA[0][ar][ac]), lds_addr(&lA[1][ar][ac]) };
  const unsigned lb[2] = { lds_addr(&lB[0][br][0]),  lds_addr(&lB[1][br][0]) };

  v8f acc[4][4] = {};

  // prologue: DMA first K-chunk into buffer 0
  async_ld32B(la[0], ga);
  async_ld64B(lb[0], gb);

  for (int k0 = 0; k0 < K; k0 += 32) {
    const int cur = (k0 >> 5) & 1;
    wait_async0();          // this thread's tile data landed in LDS
    __syncthreads();        // whole block's data landed; old buffer free

    if (k0 + 32 < K) {      // DMA next chunk into the other buffer (overlap)
      async_ld32B(la[cur ^ 1], ga + k0 + 32);
      async_ld64B(lb[cur ^ 1], gb + k0 + 32);
      __builtin_prefetch(ga + k0 + 64, 0, 1);   // warm GL2 one chunk ahead
      __builtin_prefetch(gb + k0 + 64, 0, 1);
    }

    v16h af[4], bf[4];
#pragma unroll
    for (int i = 0; i < 4; ++i)
      af[i] = load_afrag(&lA[cur][wm + i * 16][0], 32, lane);
#pragma unroll
    for (int j = 0; j < 4; ++j)
      bf[j] = load_bfrag(&lB[cur][wn + j * 16][0], 32, lane);
#pragma unroll
    for (int i = 0; i < 4; ++i) {
      // j>0 reuses the A operand (OPSEL.RA hint); hint is a constant template arg
      acc[i][0] = wmma_f16<false>(af[i], bf[0], acc[i][0]);
      acc[i][1] = wmma_f16<true >(af[i], bf[1], acc[i][1]);
      acc[i][2] = wmma_f16<true >(af[i], bf[2], acc[i][2]);
      acc[i][3] = wmma_f16<true >(af[i], bf[3], acc[i][3]);
    }
  }

  // ---- epilogue: C layout -> global f32 ----
  const int nloc = lane & 15;
  const int mhal = ((lane >> 4) & 1) * 8;
#pragma unroll
  for (int i = 0; i < 4; ++i)
#pragma unroll
    for (int j = 0; j < 4; ++j) {
      const int ncol = tile_n + wn + j * 16 + nloc;
      const int mrow = tile_m + wm + i * 16 + mhal;
#pragma unroll
      for (int v = 0; v < 8; ++v)
        C[(size_t)(mrow + v) * N + ncol] = acc[i][j][v];
    }
}

// ---------------------------------------------------------------------------
// Fused per-head RMS-norm + RoPE + f16 cast.  Q is pre-scaled by HD^-0.5
// (RoPE is linear, so this is exact) to remove score scaling from the
// attention hot loop.
// One wave per (token, slot); slots: 0..31 Q heads, 32..39 K heads, 40..47 V.
// Outputs: Q [B,S,NH,HD], K [B,NKV,S,HD], V transposed [B,NKV,HD,S].
// ---------------------------------------------------------------------------
__global__ __launch_bounds__(256)
void norm_rope_f16(const float* __restrict__ qkv,
                   const int* __restrict__ positions,
                   const float* __restrict__ qw,
                   const float* __restrict__ kw,
                   _Float16* __restrict__ qh,
                   _Float16* __restrict__ kh,
                   _Float16* __restrict__ vh) {
  const int gid  = blockIdx.x * 8 + (threadIdx.x >> 5);
  const int lane = threadIdx.x & 31;
  const int slot = gid % 48;
  const int tok  = gid / 48;
  const int b = tok >> 10, s = tok & 1023;

  const bool isQ = slot < 32;
  const bool isV = slot >= 40;
  const int col0 = isQ ? slot * HDc
                       : (isV ? 5120 + (slot - 40) * HDc
                              : 4096 + (slot - 32) * HDc);

  const float* src = qkv + (size_t)tok * QKVN + col0;
  float x[4];
#pragma unroll
  for (int i = 0; i < 4; ++i) x[i] = src[lane * 4 + i];

  if (isV) {  // plain cast, layout [B,NKV,HD,S]
    _Float16* dst = vh + (((size_t)(b * NKVc + (slot - 40)) * HDc)) * Sc;
#pragma unroll
    for (int i = 0; i < 4; ++i)
      dst[(size_t)(lane * 4 + i) * Sc + s] = (_Float16)x[i];
    return;
  }

  // RMS norm over HD=128 (full-wave reduction, wave32)
  float ss = x[0]*x[0] + x[1]*x[1] + x[2]*x[2] + x[3]*x[3];
#pragma unroll
  for (int off = 16; off >= 1; off >>= 1) ss += __shfl_xor(ss, off, 32);
  const float r = rsqrtf(ss * (1.0f / HDc) + 1e-6f);
  const float* w = isQ ? qw : kw;
#pragma unroll
  for (int i = 0; i < 4; ++i) x[i] = x[i] * r * w[lane * 4 + i];

  // RoPE: partner element at d ^ 64 lives in lane ^ 16
  float xp[4];
#pragma unroll
  for (int i = 0; i < 4; ++i) xp[i] = __shfl_xor(x[i], 16, 32);
  const bool hi = lane >= 16;
  const float pos = (float)positions[b * Sc + s];
  const float L = 13.287712379549449f / 64.0f; // log2(10000)/64
  const float oscale = isQ ? SCALEc : 1.0f;
  float y[4];
#pragma unroll
  for (int i = 0; i < 4; ++i) {
    const int d  = lane * 4 + i;
    const int fi = hi ? d - 64 : d;
    const float ang = pos * exp2f(-(float)fi * L);
    const float c = __cosf(ang), sn = __sinf(ang);
    y[i] = (hi ? (x[i] * c + xp[i] * sn) : (x[i] * c - xp[i] * sn)) * oscale;
  }

  if (isQ) {
    _Float16* dst = qh + ((size_t)tok * NHc + slot) * HDc;
#pragma unroll
    for (int i = 0; i < 4; ++i) dst[lane * 4 + i] = (_Float16)y[i];
  } else {
    _Float16* dst = kh + (((size_t)(b * NKVc + (slot - 32)) * Sc + s)) * HDc;
#pragma unroll
    for (int i = 0; i < 4; ++i) dst[lane * 4 + i] = (_Float16)y[i];
  }
}

// ---------------------------------------------------------------------------
// Flash attention (causal, GQA 4:1).  Block = 8 waves x 16 q-rows = 128 rows.
// Key blocks of 32: QK^T = 8 WMMAs, online softmax, PV = 8 WMMAs.
// K/V tiles double-buffered via async-to-LDS (ASYNCcnt pipelining).
// ---------------------------------------------------------------------------
__global__ __launch_bounds__(256)
void flash_attn_wmma(const _Float16* __restrict__ qh,
                     const _Float16* __restrict__ kh,
                     const _Float16* __restrict__ vh,
                     _Float16* __restrict__ oh) {
  const int qblk = blockIdx.x, h = blockIdx.y, b = blockIdx.z;
  const int kvh  = h >> 2;                  // NH/NKV = 4
  const int tid  = threadIdx.x;
  const int lane = tid & 31;
  const int wave = tid >> 5;
  const int qrow0 = qblk * 128 + wave * 16;

  __shared__ alignas(16) _Float16 kt [2][32][128];   // [key][hd]
  __shared__ alignas(16) _Float16 vtt[2][128][32];   // [hd][key]
  __shared__ alignas(16) _Float16 pt [8][16][32];    // per-wave P tile

  const _Float16* kbase = kh + (size_t)(b * NKVc + kvh) * Sc * HDc;
  const _Float16* vbase = vh + (size_t)(b * NKVc + kvh) * HDc * Sc;
  const _Float16* qbase = qh + (((size_t)(b * Sc + qrow0) * NHc + h) * HDc);

  // staging addresses (32B per thread per tile)
  const int kr = tid >> 3, kc = (tid & 7) * 16;   // K tile [32][128]
  const int vr = tid >> 1, vc = (tid & 1) * 16;   // V tile [128][32]
  const _Float16* gk = kbase + (size_t)kr * HDc + kc;
  const _Float16* gv = vbase + (size_t)vr * Sc + vc;
  const unsigned lk[2] = { lds_addr(&kt[0][kr][kc]),  lds_addr(&kt[1][kr][kc]) };
  const unsigned lv[2] = { lds_addr(&vtt[0][vr][vc]), lds_addr(&vtt[1][vr][vc]) };

  // Q fragments (held in registers for the whole key loop)
  v16h qf[4];
#pragma unroll
  for (int c = 0; c < 4; ++c)
    qf[c] = load_afrag(qbase + c * 32, NHc * HDc, lane);

  v8f  o[8] = {};
  float mrow[8], lrow[8];
#pragma unroll
  for (int v = 0; v < 8; ++v) { mrow[v] = -1e30f; lrow[v] = 0.0f; }

  const int kend = qblk * 128 + 128;

  // prologue: DMA first key block into buffer 0
  async_ld32B(lk[0], gk);
  async_ld32B(lv[0], gv);

  for (int kb0 = 0; kb0 < kend; kb0 += 32) {
    const int cur = (kb0 >> 5) & 1;
    wait_async0();
    __syncthreads();        // current K/V tile ready; old buffer free

    if (kb0 + 32 < kend) {  // DMA next key block while computing this one
      async_ld32B(lk[cur ^ 1], gk + (size_t)(kb0 + 32) * HDc);
      async_ld32B(lv[cur ^ 1], gv + (kb0 + 32));
    }

    // ---- scores: two 16-key C tiles, contraction over HD in 4 chunks ----
    v8f c0 = {}, c1 = {};
#pragma unroll
    for (int c = 0; c < 4; ++c) {
      v16h kf0 = load_bfrag(&kt[cur][0][c * 32],  HDc, lane);
      v16h kf1 = load_bfrag(&kt[cur][16][c * 32], HDc, lane);
      c0 = wmma_f16<false>(qf[c], kf0, c0);
      // same A operand as the previous WMMA -> reuse hint
      c1 = wmma_f16<true >(qf[c], kf1, c1);
    }

    // ---- online softmax (per row; rows live in 16-lane halves) ----
    const int nloc = lane & 15;
    const int mhal = ((lane >> 4) & 1) * 8;
#pragma unroll
    for (int v = 0; v < 8; ++v) {
      const int m = qrow0 + v + mhal;
      float s0 = c0[v];                 // scale pre-folded into Q
      float s1 = c1[v];
      if (kb0 + nloc      > m) s0 = -1e30f;
      if (kb0 + 16 + nloc > m) s1 = -1e30f;
      float mx = fmaxf(s0, s1);
#pragma unroll
      for (int off = 8; off >= 1; off >>= 1)
        mx = fmaxf(mx, __shfl_xor(mx, off, 32));
      const float mnew  = fmaxf(mrow[v], mx);
      const float alpha = __expf(mrow[v] - mnew);
      const float p0 = __expf(s0 - mnew);
      const float p1 = __expf(s1 - mnew);
      float rs = p0 + p1;
#pragma unroll
      for (int off = 8; off >= 1; off >>= 1) rs += __shfl_xor(rs, off, 32);
      lrow[v] = lrow[v] * alpha + rs;
      mrow[v] = mnew;
#pragma unroll
      for (int j = 0; j < 8; ++j) o[j][v] *= alpha;
      pt[wave][v + mhal][nloc]      = (_Float16)p0;
      pt[wave][v + mhal][nloc + 16] = (_Float16)p1;
    }
    __syncthreads();  // P tile visible across the wave's lanes

    // ---- PV: P(16x32) x V(32x128) -> 8 HD tiles (P reused 8x) ----
    v16h pf = load_afrag(&pt[wave][0][0], 32, lane);
    o[0] = wmma_f16<false>(pf, load_bfrag(&vtt[cur][0][0], 32, lane), o[0]);
#pragma unroll
    for (int j = 1; j < 8; ++j) {
      v16h vf = load_bfrag(&vtt[cur][j * 16][0], 32, lane);
      o[j] = wmma_f16<true>(pf, vf, o[j]);
    }
  }

  // ---- normalize + store [B,S,NH,HD] as f16 ----
  const int nloc = lane & 15;
  const int mhal = ((lane >> 4) & 1) * 8;
  float inv[8];
#pragma unroll
  for (int v = 0; v < 8; ++v) inv[v] = 1.0f / lrow[v];
#pragma unroll
  for (int v = 0; v < 8; ++v) {
    const int srow = qrow0 + v + mhal;
    _Float16* dst = oh + (((size_t)(b * Sc + srow) * NHc + h) * HDc);
#pragma unroll
    for (int j = 0; j < 8; ++j)
      dst[j * 16 + nloc] = (_Float16)(o[j][v] * inv[v]);
  }
}

// ---------------------------------------------------------------------------
// Launcher
// ---------------------------------------------------------------------------
extern "C" void kernel_launch(void* const* d_in, const int* in_sizes, int n_in,
                              void* d_out, int out_size, void* d_ws, size_t ws_size,
                              hipStream_t stream) {
  const float* hs   = (const float*)d_in[0];
  const int*   pos  = (const int*)d_in[1];
  const float* wqkv = (const float*)d_in[2];
  const float* wout = (const float*)d_in[3];
  const float* qnw  = (const float*)d_in[4];
  const float* knw  = (const float*)d_in[5];
  float* out = (float*)d_out;

  char* w = (char*)d_ws;
  _Float16* hs_h   = (_Float16*)w; w += (size_t)Bc*Sc*Hc      * 2;
  _Float16* wqkvT  = (_Float16*)w; w += (size_t)Hc*QKVN       * 2;  // [6144][4096]
  _Float16* woutT  = (_Float16*)w; w += (size_t)Hc*Hc         * 2;  // [4096][4096]
  float*    qkv    = (float*)w;    w += (size_t)Bc*Sc*QKVN    * 4;
  _Float16* qh     = (_Float16*)w; w += (size_t)Bc*Sc*NHc*HDc * 2;
  _Float16* kh     = (_Float16*)w; w += (size_t)Bc*NKVc*Sc*HDc* 2;
  _Float16* vh     = (_Float16*)w; w += (size_t)Bc*NKVc*Sc*HDc* 2;
  _Float16* oh     = (_Float16*)w; w += (size_t)Bc*Sc*NHc*HDc * 2;
  (void)n_in; (void)in_sizes; (void)out_size; (void)ws_size;

  const int nh = Bc*Sc*Hc;
  cvt_f32_f16<<<nh / 1024, 256, 0, stream>>>(hs, hs_h, nh);
  // weights: transpose-convert so GEMM B tiles stage as pure b128 async copies
  transpose_cvt_f32_f16<<<dim3(QKVN / 32, Hc / 32), 256, 0, stream>>>(
      wqkv, wqkvT, Hc, QKVN);
  transpose_cvt_f32_f16<<<dim3(Hc / 32, Hc / 32), 256, 0, stream>>>(
      wout, woutT, Hc, Hc);

  // QKV projection: [2048,4096] x [4096,6144] -> f32
  gemm_f16_wmma<<<dim3(QKVN / 256, (Bc*Sc) / 128), 256, 0, stream>>>(
      hs_h, wqkvT, qkv, Bc*Sc, QKVN, Hc);

  // RMS-norm + RoPE + layout transforms (Q pre-scaled by HD^-0.5)
  norm_rope_f16<<<(Bc*Sc*48) / 8, 256, 0, stream>>>(
      qkv, pos, qnw, knw, qh, kh, vh);

  // Flash attention
  flash_attn_wmma<<<dim3(Sc / 128, NHc, Bc), 256, 0, stream>>>(qh, kh, vh, oh);

  // Output projection: [2048,4096] x [4096,4096] -> f32 out
  gemm_f16_wmma<<<dim3(Hc / 256, (Bc*Sc) / 128), 256, 0, stream>>>(
      oh, woutT, out, Bc*Sc, Hc, Hc);
}
